// LSTMCell_29944511987933
// MI455X (gfx1250) — matrix operand order repaired and checked by reading, not compiled, not run on hardware
//
#include <hip/hip_runtime.h>
#include <hip/hip_bf16.h>
#include <math.h>

typedef __bf16 bf16_t;
typedef __bf16 v16bf __attribute__((ext_vector_type(16)));
typedef float  v8f   __attribute__((ext_vector_type(8)));

union Frag { uint4 u[2]; v16bf v; };

#define LDT 72   // 64 halfs per tile row + 8 halfs pad (keeps 16B alignment, spreads banks)

// batched pointer set so 4 gate GEMMs run in one launch (gridDim.z)
struct GemmBatch {
    const bf16_t* Bt[4];
    const float*  bias[4];
    float*        C[4];
};

// ---------------------------------------------------------------------------
// A = [x | h] concatenated along K, converted to bf16. A is (2048 x 4096).
// ---------------------------------------------------------------------------
__global__ void cat_bf16_kernel(const float* __restrict__ x,
                                const float* __restrict__ h,
                                bf16_t* __restrict__ A) {
    int idx = blockIdx.x * 256 + threadIdx.x;     // < 2048*2048
    int m = idx >> 11;
    int j = idx & 2047;
    A[(size_t)m * 4096 + j]        = (bf16_t)x[idx];
    A[(size_t)m * 4096 + 2048 + j] = (bf16_t)h[idx];
}

// ---------------------------------------------------------------------------
// Transpose + f32->bf16 convert: W (K x N, row major) -> Wt (Npad x ldWt,
// N-major, written at column offset koff). Rows n >= N are zero-filled.
// block(32,8), grid(Npad/32, K/32).
// ---------------------------------------------------------------------------
__global__ void transpose_cvt_kernel(const float* __restrict__ W,
                                     int K, int N, int Npad,
                                     bf16_t* __restrict__ Wt, int ldWt, int koff) {
    __shared__ bf16_t tile[32][33];
    int nt = blockIdx.x * 32, kt = blockIdx.y * 32;
    int tx = threadIdx.x, ty = threadIdx.y;
#pragma unroll
    for (int i = 0; i < 4; ++i) {
        int k = kt + ty + i * 8;
        int n = nt + tx;
        float v = (n < N) ? W[(size_t)k * N + n] : 0.f;
        tile[ty + i * 8][tx] = (bf16_t)v;
    }
    __syncthreads();
#pragma unroll
    for (int i = 0; i < 4; ++i) {
        int n = nt + ty + i * 8;
        int k = kt + tx;
        if (n < Npad) Wt[(size_t)n * ldWt + koff + k] = tile[tx][ty + i * 8];
    }
}

// ---------------------------------------------------------------------------
// C(MxN, f32) = A(MxK, bf16 row-major) * Bt(NxK, bf16 N-major)^T + bias
// Block tile 128x256, BK=64. 256 threads = 8 wave32 waves, wave grid 2(M) x
// 4(N); each wave computes a 64x64 tile = 4x4 WMMA 16x16 accumulators
// (1.0 ds_load_b128 per v_wmma in steady state). Global->LDS staging uses
// gfx1250 async copies (global_load_async_to_lds_b128, ASYNCcnt), double
// buffered so the next K-tile streams in during compute. gridDim.z selects
// the weight/bias/output set (4 fused gate GEMMs share A via L2).
// ---------------------------------------------------------------------------
__global__ __launch_bounds__(256) void gemm_bf16_wmma(
    const bf16_t* __restrict__ A, GemmBatch batch, int biasN,
    int M, int N, int K) {

    const bf16_t* __restrict__ Bt   = batch.Bt[blockIdx.z];
    const float*  __restrict__ bias = batch.bias[blockIdx.z];
    float*        __restrict__ C    = batch.C[blockIdx.z];

    __shared__ __align__(16) bf16_t sA[2][128 * LDT];
    __shared__ __align__(16) bf16_t sB[2][256 * LDT];

    const int tid    = threadIdx.x;
    const int lane   = tid & 31;
    const int wid    = tid >> 5;
    const int bm     = blockIdx.y * 128;
    const int bn     = blockIdx.x * 256;
    const int wmOff  = (wid & 1) * 64;   // wave offset in M (2 waves)
    const int wnOff  = (wid >> 1) * 64;  // wave offset in N (4 waves)
    const int mrow   = lane & 15;        // M row (A) or N col (B,C) within tile
    const int laneHi = lane >> 4;        // upper/lower 16-lane half

    v8f acc[4][4];
#pragma unroll
    for (int mi = 0; mi < 4; ++mi)
#pragma unroll
        for (int ni = 0; ni < 4; ++ni)
            acc[mi][ni] = {};

    // issue one (128x64 A + 256x64 B) bf16 tile pair into LDS buffer `buf`:
    // 4 + 8 = 12 async 16B copies per thread -> ASYNCcnt += 12 per thread
    auto issue_tile = [&](int buf, int kt) {
#pragma unroll
        for (int i = 0; i < 4; ++i) {
            int idx = tid + i * 256;       // 0..1023
            int row = idx >> 3;            // 0..127
            int ch  = (idx & 7) * 8;       // half offset within the 64-wide row
            unsigned la = (unsigned)(uintptr_t)&sA[buf][row * LDT + ch];
            unsigned long long ga =
                (unsigned long long)(uintptr_t)(A + (size_t)(bm + row) * K + kt + ch);
            asm volatile("global_load_async_to_lds_b128 %0, %1, off"
                         :: "v"(la), "v"(ga) : "memory");
        }
#pragma unroll
        for (int i = 0; i < 8; ++i) {
            int idx = tid + i * 256;       // 0..2047
            int row = idx >> 3;            // 0..255
            int ch  = (idx & 7) * 8;
            unsigned lb = (unsigned)(uintptr_t)&sB[buf][row * LDT + ch];
            unsigned long long gb =
                (unsigned long long)(uintptr_t)(Bt + (size_t)(bn + row) * K + kt + ch);
            asm volatile("global_load_async_to_lds_b128 %0, %1, off"
                         :: "v"(lb), "v"(gb) : "memory");
        }
    };

    int cur = 0;
    issue_tile(0, 0);

    for (int kt = 0; kt < K; kt += 64) {
        const bool hasNext = (kt + 64 < K);
        if (hasNext) {
            issue_tile(cur ^ 1, kt + 64);
            // 24 outstanding per thread; async loads complete in order, so
            // waiting down to 12 drains exactly the current tile's batch.
            asm volatile("s_wait_asynccnt 12" ::: "memory");
        } else {
            asm volatile("s_wait_asynccnt 0" ::: "memory");
        }
        __syncthreads();   // publish this tile's LDS data to all waves

        const bf16_t* cA = sA[cur];
        const bf16_t* cB = sB[cur];
#pragma unroll
        for (int kk = 0; kk < 64; kk += 32) {
            // A fragment (16x32 bf16): lane mrow, K halves {8*laneHi..+7} and {16+8*laneHi..+7}
            Frag a[4];
#pragma unroll
            for (int mi = 0; mi < 4; ++mi) {
                const bf16_t* base = &cA[(wmOff + mi * 16 + mrow) * LDT + kk + laneHi * 8];
                a[mi].u[0] = *(const uint4*)(base);
                a[mi].u[1] = *(const uint4*)(base + 16);
            }
            // B fragment (32x16 bf16): lane col mrow, K halves {8*laneHi..+7} and {16+8*laneHi..+7}
            Frag b[4];
#pragma unroll
            for (int ni = 0; ni < 4; ++ni) {
                const bf16_t* base = &cB[(wnOff + ni * 16 + mrow) * LDT + kk + laneHi * 8];
                b[ni].u[0] = *(const uint4*)(base);
                b[ni].u[1] = *(const uint4*)(base + 16);
            }
#pragma unroll
            for (int mi = 0; mi < 4; ++mi)
#pragma unroll
                for (int ni = 0; ni < 4; ++ni)
                    acc[mi][ni] = __builtin_amdgcn_wmma_f32_16x16x32_bf16(
                        false, a[mi].v, false, b[ni].v,
                        (short)0, acc[mi][ni], false, false);
        }
        __syncthreads();   // all waves done reading before buffer is re-filled
        cur ^= 1;
    }

    // ---- epilogue: C/D layout -> lane mrow = N col, VGPR v = M row v+8*laneHi
#pragma unroll
    for (int ni = 0; ni < 4; ++ni) {
        int col  = bn + wnOff + ni * 16 + mrow;
        float bv = (col < biasN) ? bias[col] : 0.f;
#pragma unroll
        for (int mi = 0; mi < 4; ++mi) {
#pragma unroll
            for (int v = 0; v < 8; ++v) {
                int row = bm + wmOff + mi * 16 + laneHi * 8 + v;
                C[(size_t)row * N + col] = acc[mi][ni][v] + bv;
            }
        }
    }
}

// ---------------------------------------------------------------------------
// Pointwise LSTM gate math.
// ---------------------------------------------------------------------------
__global__ void lstm_pointwise_kernel(const float* __restrict__ pg,
                                      const float* __restrict__ pi,
                                      const float* __restrict__ pf,
                                      const float* __restrict__ po,
                                      const float* __restrict__ c,
                                      float* __restrict__ c_out,
                                      float* __restrict__ h_out,
                                      bf16_t* __restrict__ hb) {
    int idx = blockIdx.x * 256 + threadIdx.x;
    float g = tanhf(pg[idx]);
    float i = 1.f / (1.f + __expf(-pi[idx]));
    float f = 1.f / (1.f + __expf(-pf[idx]));
    float o = 1.f / (1.f + __expf(-po[idx]));
    float cn = g * i + c[idx] * f;
    float hn = tanhf(cn) * o;
    c_out[idx] = cn;
    h_out[idx] = hn;
    hb[idx]    = (bf16_t)hn;
}

// ---------------------------------------------------------------------------
// Row softmax over ncls columns (ldp-padded input).
// ---------------------------------------------------------------------------
__global__ void softmax_kernel(const float* __restrict__ P, int ldp, int ncls,
                               float* __restrict__ Y) {
    int row = blockIdx.x;
    const float* p = P + (size_t)row * ldp;
    __shared__ float red[256];
    int tid = threadIdx.x;

    float m = -INFINITY;
    for (int j = tid; j < ncls; j += 256) m = fmaxf(m, p[j]);
    red[tid] = m; __syncthreads();
    for (int s = 128; s > 0; s >>= 1) {
        if (tid < s) red[tid] = fmaxf(red[tid], red[tid + s]);
        __syncthreads();
    }
    m = red[0]; __syncthreads();

    float sum = 0.f;
    for (int j = tid; j < ncls; j += 256) sum += __expf(p[j] - m);
    red[tid] = sum; __syncthreads();
    for (int s = 128; s > 0; s >>= 1) {
        if (tid < s) red[tid] += red[tid + s];
        __syncthreads();
    }
    float inv = 1.f / red[0];
    for (int j = tid; j < ncls; j += 256)
        Y[(size_t)row * ncls + j] = __expf(p[j] - m) * inv;
}

// ---------------------------------------------------------------------------
extern "C" void kernel_launch(void* const* d_in, const int* in_sizes, int n_in,
                              void* d_out, int out_size, void* d_ws, size_t ws_size,
                              hipStream_t stream) {
    constexpr int B = 2048, D = 2048, KCAT = 4096, NCLS = 1000, NPAD = 1024;

    const float* x   = (const float*)d_in[0];
    const float* c   = (const float*)d_in[1];
    const float* h   = (const float*)d_in[2];
    const float* Wx[4] = { (const float*)d_in[3], (const float*)d_in[4],
                           (const float*)d_in[5], (const float*)d_in[6] };   // g,i,f,o
    const float* Wh[4] = { (const float*)d_in[7], (const float*)d_in[8],
                           (const float*)d_in[9], (const float*)d_in[10] };
    const float* bias[4] = { (const float*)d_in[11], (const float*)d_in[12],
                             (const float*)d_in[13], (const float*)d_in[14] };
    const float* Wph = (const float*)d_in[15];
    const float* bp  = (const float*)d_in[16];

    // workspace layout
    char* ws = (char*)d_ws;
    size_t off = 0;
    bf16_t* Acat = (bf16_t*)(ws + off); off += (size_t)B * KCAT * 2;          // 16.8 MB
    bf16_t* Wt[4];
    for (int gi = 0; gi < 4; ++gi) { Wt[gi] = (bf16_t*)(ws + off); off += (size_t)D * KCAT * 2; }
    bf16_t* WphT = (bf16_t*)(ws + off); off += (size_t)NPAD * D * 2;          // 4.2 MB
    float* pre[4];
    for (int gi = 0; gi < 4; ++gi) { pre[gi] = (float*)(ws + off); off += (size_t)B * D * 4; }
    bf16_t* hb = (bf16_t*)(ws + off); off += (size_t)B * D * 2;               // 8.4 MB
    float* pfull = (float*)(ws + off); off += (size_t)B * NPAD * 4;           // 8.4 MB

    float* y     = (float*)d_out;
    float* c_out = y + (size_t)B * NCLS;
    float* h_out = c_out + (size_t)B * D;

    // 1) A = [x | h] in bf16
    cat_bf16_kernel<<<(B * D) / 256, 256, 0, stream>>>(x, h, Acat);

    // 2) weight transpose/convert: Wt[g] = [Wx[g]^T | Wh[g]^T]  (D x 4096)
    for (int gi = 0; gi < 4; ++gi) {
        transpose_cvt_kernel<<<dim3(D / 32, D / 32), dim3(32, 8), 0, stream>>>(
            Wx[gi], D, D, D, Wt[gi], KCAT, 0);
        transpose_cvt_kernel<<<dim3(D / 32, D / 32), dim3(32, 8), 0, stream>>>(
            Wh[gi], D, D, D, Wt[gi], KCAT, D);
    }
    // W_ph^T padded to 1024 rows
    transpose_cvt_kernel<<<dim3(NPAD / 32, D / 32), dim3(32, 8), 0, stream>>>(
        Wph, D, NCLS, NPAD, WphT, D, 0);

    // 3) four fused gate GEMMs (gridDim.z): pre[g] = A @ [Wx;Wh] + b
    {
        GemmBatch gates;
        for (int gi = 0; gi < 4; ++gi) {
            gates.Bt[gi]   = Wt[gi];
            gates.bias[gi] = bias[gi];
            gates.C[gi]    = pre[gi];
        }
        gemm_bf16_wmma<<<dim3(D / 256, B / 128, 4), 256, 0, stream>>>(
            Acat, gates, D, B, D, KCAT);
    }

    // 4) gate nonlinearity + cell/hidden update
    lstm_pointwise_kernel<<<(B * D) / 256, 256, 0, stream>>>(
        pre[0], pre[1], pre[2], pre[3], c, c_out, h_out, hb);

    // 5) projection: p = h_new @ W_ph + b_p (N padded to 1024, K=2048)
    {
        GemmBatch proj;
        proj.Bt[0] = WphT; proj.bias[0] = bp; proj.C[0] = pfull;
        for (int gi = 1; gi < 4; ++gi) { proj.Bt[gi] = WphT; proj.bias[gi] = bp; proj.C[gi] = pfull; }
        gemm_bf16_wmma<<<dim3(NPAD / 256, B / 128, 1), 256, 0, stream>>>(
            hb, proj, NCLS, B, NPAD, D);
    }

    // 6) softmax over the first 1000 columns
    softmax_kernel<<<B, 256, 0, stream>>>(pfull, NPAD, NCLS, y);
}
